// MaskAttn2_56607668961918
// MI455X (gfx1250) — matrix operand order repaired and checked by reading, compiled, not run on hardware
//
#include <hip/hip_runtime.h>
#include <stdint.h>

#define BB 4
#define NN 2048
#define CC 1024
#define HH 16
#define HD 64

typedef __bf16 bf16_t;
typedef __attribute__((ext_vector_type(16))) __bf16 v16bf;
typedef __attribute__((ext_vector_type(8)))  __bf16 v8bf;
typedef __attribute__((ext_vector_type(8)))  float  v8f;

union V16 { v16bf v; v8bf h[2]; };

static __device__ __forceinline__ bf16_t f2bf(float f) {
  unsigned u = __builtin_bit_cast(unsigned, f);
  unsigned r = (u + 0x7fffu + ((u >> 16) & 1u)) >> 16;
  unsigned short s = (unsigned short)r;
  return __builtin_bit_cast(bf16_t, s);
}

// A fragment, 16x32 bf16, row-major source with leading dim lda (elements).
// lanes 0-15: rows 0-15, k = {0..7, 16..23}; lanes 16-31: k = {8..15, 24..31}.
static __device__ __forceinline__ v16bf load_a_frag(const bf16_t* A, int lda, int lane) {
  int r  = lane & 15;
  int kb = (lane >> 4) << 3;
  V16 u;
  u.h[0] = *(const v8bf*)(A + (size_t)r * lda + kb);
  u.h[1] = *(const v8bf*)(A + (size_t)r * lda + kb + 16);
  return u.v;
}

// B fragment, 32x16 bf16: element (k,n) = Wc[n*ldw + k]; per lane col n=lane&15,
// lanes 0-15: k = {0..7, 16..23}; lanes 16-31: k = {8..15, 24..31}.
static __device__ __forceinline__ v16bf load_b_frag(const bf16_t* Wc, int ldw, int lane) {
  int n  = lane & 15;
  int kb = (lane >> 4) << 3;
  V16 u;
  u.h[0] = *(const v8bf*)(Wc + (size_t)n * ldw + kb);
  u.h[1] = *(const v8bf*)(Wc + (size_t)n * ldw + kb + 16);
  return u.v;
}

static __device__ __forceinline__ v8f wmma_bf16(v16bf a, v16bf b, v8f c) {
  return __builtin_amdgcn_wmma_f32_16x16x32_bf16(false, a, false, b, (short)0, c, false, false);
}

// Async DMA of 16 bytes (per lane) from global into LDS, tracked by ASYNCcnt.
// Low 32 bits of a flat LDS address are the wave-relative LDS byte offset
// (ISA 10.2).
static __device__ __forceinline__ void async_tile16(const bf16_t* gp, bf16_t* lp) {
  unsigned ldsa = (unsigned)(size_t)(void*)lp;
  unsigned long long ga = (unsigned long long)(size_t)(const void*)gp;
  asm volatile("global_load_async_to_lds_b128 %0, %1, off"
               :: "v"(ldsa), "v"(ga) : "memory");
}

#define WAIT_ASYNC_0() asm volatile("s_wait_asynccnt 0x0" ::: "memory")
#define WAIT_ASYNC_1() asm volatile("s_wait_asynccnt 0x1" ::: "memory")
#define WAIT_ASYNC_2() asm volatile("s_wait_asynccnt 0x2" ::: "memory")

#define SWZ(v, mask) __builtin_bit_cast(float, __builtin_amdgcn_ds_swizzle( \
    __builtin_bit_cast(int, (v)), (((mask) << 10) | 0x1f)))

// reductions across each 16-lane half (rows of a WMMA C fragment)
static __device__ __forceinline__ float gmax(float v) {
  v = fmaxf(v, SWZ(v, 1)); v = fmaxf(v, SWZ(v, 2));
  v = fmaxf(v, SWZ(v, 4)); v = fmaxf(v, SWZ(v, 8));
  return v;
}
static __device__ __forceinline__ float gsum(float v) {
  v += SWZ(v, 1); v += SWZ(v, 2); v += SWZ(v, 4); v += SWZ(v, 8);
  return v;
}

// ---------------- prep kernels ----------------

__global__ __launch_bounds__(256) void k_f2bf(const float* __restrict__ src,
                                              bf16_t* __restrict__ dst, int n) {
  for (int i = blockIdx.x * blockDim.x + threadIdx.x; i < n; i += gridDim.x * blockDim.x)
    dst[i] = f2bf(src[i]);
}

__global__ __launch_bounds__(256) void k_masksum(const float* __restrict__ mWin,
                                                 float* __restrict__ msum) {
  __shared__ float red[256];
  int b = blockIdx.x;
  float s = 0.0f;
  for (int i = threadIdx.x; i < NN; i += 256) {
    float m = mWin[b * NN + i];
    s += m * m;
  }
  red[threadIdx.x] = s;
  __syncthreads();
  for (int off = 128; off > 0; off >>= 1) {
    if ((int)threadIdx.x < off) red[threadIdx.x] += red[threadIdx.x + off];
    __syncthreads();
  }
  if (threadIdx.x == 0) msum[b] = red[0];
}

// ---------------- QKV GEMM: [8192 x 1024] @ W^T -> masked q/k/v bf16 ----------------
// block tile 256(M) x 64(N); 8 waves, each 32 rows x 4 col tiles (8 WMMA accs).
// W tile (64 cols x 32 k) double-buffered in LDS via async-to-LDS DMA.
// Last two K-iterations are peeled so the steady-state loop is branch-free.

__global__ __launch_bounds__(256) void k_qkv(const bf16_t* __restrict__ xb,
                                             const bf16_t* __restrict__ wq,
                                             const float* __restrict__ mWin,
                                             bf16_t* __restrict__ qb,
                                             bf16_t* __restrict__ kbuf,
                                             bf16_t* __restrict__ vt) {
  __shared__ bf16_t Bs[2][64 * 32];
  int colBase = blockIdx.x * 64;   // over 3072
  int rowBase = blockIdx.y * 256;  // over 8192
  int wave = threadIdx.x >> 5, lane = threadIdx.x & 31;
  int waveRow = rowBase + wave * 32;
  int hi8 = (lane >> 4) << 3;
  int ncol = lane & 15;

  v8f acc[2][4] = {{{}, {}, {}, {}}, {{}, {}, {}, {}}};

  // cooperative staging map: thread -> one 16B chunk of the 64x32 W tile
  int srow = threadIdx.x >> 2;
  int sch  = (threadIdx.x & 3) * 8;
  const bf16_t* wsrc = wq + (size_t)(colBase + srow) * CC + sch;
  bf16_t* ldst0 = &Bs[0][srow * 32 + sch];
  bf16_t* ldst1 = &Bs[1][srow * 32 + sch];

  async_tile16(wsrc + 0,  ldst0);   // tile 0
  async_tile16(wsrc + 32, ldst1);   // tile 1

  auto step = [&](int k0, int buf) {
    v16bf a0 = load_a_frag(xb + (size_t)waveRow * CC + k0, CC, lane);
    v16bf a1 = load_a_frag(xb + (size_t)(waveRow + 16) * CC + k0, CC, lane);
#pragma unroll
    for (int t = 0; t < 4; ++t) {
      v16bf bfrag = load_b_frag(&Bs[buf][(t * 16) * 32], 32, lane);
      acc[0][t] = wmma_bf16(a0, bfrag, acc[0][t]);
      acc[1][t] = wmma_bf16(a1, bfrag, acc[1][t]);
    }
  };

  for (int k0 = 0; k0 < CC - 64; k0 += 32) {
    int buf = (k0 >> 5) & 1;
    WAIT_ASYNC_1();
    __syncthreads();
    step(k0, buf);
    __syncthreads();
    async_tile16(wsrc + k0 + 64, buf ? ldst1 : ldst0);
  }
  WAIT_ASYNC_1();               // tile CC-64 done (only CC-32 outstanding)
  __syncthreads();
  step(CC - 64, ((CC - 64) >> 5) & 1);
  WAIT_ASYNC_0();               // drain
  __syncthreads();
  step(CC - 32, ((CC - 32) >> 5) & 1);

#pragma unroll
  for (int mi = 0; mi < 2; ++mi) {
#pragma unroll
    for (int g = 0; g < 8; ++g) {
      int R = waveRow + mi * 16 + g + hi8;
      int b = R >> 11, tok = R & (NN - 1);
      float mw = mWin[R];
#pragma unroll
      for (int t = 0; t < 4; ++t) {
        int Cc = colBase + t * 16 + ncol;
        int which = Cc >> 10;
        int c = Cc & (CC - 1);
        int h = c >> 6, d = c & 63;
        float f = (which == 0) ? (mw * 0.125f) : mw;  // scale = hd^-0.5 = 0.125
        bf16_t v = f2bf(acc[mi][t][g] * f);
        size_t bh = (size_t)(b * HH + h);
        if (which == 0)      qb  [(bh * NN + tok) * HD + d] = v;
        else if (which == 1) kbuf[(bh * NN + tok) * HD + d] = v;
        else                 vt  [(bh * HD + d) * NN + tok] = v;  // V transposed
      }
    }
  }
}

// ---------------- flash attention ----------------
// One block = 8 waves = 128 q rows of one (b,h). K (32x64) and V^T (64x32)
// KV tiles are double-buffered in LDS via async DMA and shared by all 8 waves.
// Last two KV-iterations peeled for a branch-free steady-state loop.

__global__ __launch_bounds__(256) void k_attn(const bf16_t* __restrict__ qb,
                                              const bf16_t* __restrict__ kbuf,
                                              const bf16_t* __restrict__ vt,
                                              const float* __restrict__ mWin,
                                              const float* __restrict__ msum,
                                              bf16_t* __restrict__ ao) {
  __shared__ bf16_t Ks[2][32 * 64];  // [kv][d]
  __shared__ bf16_t Vs[2][64 * 32];  // [d][kv]
  __shared__ bf16_t Pl[8][16 * 32];  // per-wave P staging tile
  int wave = threadIdx.x >> 5, lane = threadIdx.x & 31;
  int bh = blockIdx.x >> 4;          // 64 (b,h) pairs
  int qg = blockIdx.x & 15;          // 16 q groups of 128 rows
  int q0 = qg * 128 + wave * 16;
  int b = bh >> 4, h = bh & 15;
  int hi8 = (lane >> 4) << 3;
  int ncol = lane & 15;

  const bf16_t* Q  = qb   + (size_t)bh * NN * HD + (size_t)q0 * HD;
  const bf16_t* K  = kbuf + (size_t)bh * NN * HD;
  const bf16_t* VT = vt   + (size_t)bh * HD * NN;

  v16bf aq0 = load_a_frag(Q, HD, lane);       // k = d 0..31
  v16bf aq1 = load_a_frag(Q + 32, HD, lane);  // k = d 32..63

  // staging maps: 256 threads x 16B cover each 4KB tile
  int krow = threadIdx.x >> 3, kch = (threadIdx.x & 7) * 8;  // K: 32 rows x 64
  int vrow = threadIdx.x >> 2, vch = (threadIdx.x & 3) * 8;  // V^T: 64 rows x 32
  const bf16_t* kg = K  + (size_t)krow * HD + kch;   // + j0*HD
  const bf16_t* vg = VT + (size_t)vrow * NN + vch;   // + j0
  bf16_t* kl0 = &Ks[0][krow * 64 + kch];
  bf16_t* kl1 = &Ks[1][krow * 64 + kch];
  bf16_t* vl0 = &Vs[0][vrow * 32 + vch];
  bf16_t* vl1 = &Vs[1][vrow * 32 + vch];

  async_tile16(kg, kl0);
  async_tile16(vg, vl0);
  async_tile16(kg + (size_t)32 * HD, kl1);
  async_tile16(vg + 32, vl1);

  v8f o[4] = {{}, {}, {}, {}};
  float mrun[8], lrun[8];
#pragma unroll
  for (int g = 0; g < 8; ++g) { mrun[g] = -1.0e30f; lrun[g] = 0.0f; }

  bf16_t* Pw = &Pl[wave][0];

  auto step = [&](int j0, int buf) {
    const bf16_t* Kb = &Ks[buf][0];
    v8f S0 = {}, S1 = {};
    S0 = wmma_bf16(aq0, load_b_frag(Kb, HD, lane), S0);
    S0 = wmma_bf16(aq1, load_b_frag(Kb + 32, HD, lane), S0);
    S1 = wmma_bf16(aq0, load_b_frag(Kb + 16 * HD, HD, lane), S1);
    S1 = wmma_bf16(aq1, load_b_frag(Kb + 16 * HD + 32, HD, lane), S1);

    float alpha[8];
#pragma unroll
    for (int g = 0; g < 8; ++g) {
      float mx = gmax(fmaxf(S0[g], S1[g]));
      float mn = fmaxf(mrun[g], mx);
      float al = __expf(mrun[g] - mn);
      mrun[g] = mn;
      float p0 = __expf(S0[g] - mn);
      float p1 = __expf(S1[g] - mn);
      S0[g] = p0; S1[g] = p1;
      lrun[g] = lrun[g] * al + gsum(p0 + p1);
      alpha[g] = al;
    }
#pragma unroll
    for (int t = 0; t < 4; ++t)
#pragma unroll
      for (int g = 0; g < 8; ++g) o[t][g] *= alpha[g];

    // C-layout -> A-layout via per-wave LDS tile (16 x 32, row-major)
#pragma unroll
    for (int g = 0; g < 8; ++g) {
      int m = g + hi8;
      Pw[m * 32 + ncol]      = f2bf(S0[g]);
      Pw[m * 32 + ncol + 16] = f2bf(S1[g]);
    }
    __builtin_amdgcn_wave_barrier();
    asm volatile("s_wait_dscnt 0" ::: "memory");  // LDS in-order within wave
    v16bf ap = load_a_frag(Pw, 32, lane);

    const bf16_t* Vb = &Vs[buf][0];
#pragma unroll
    for (int t = 0; t < 4; ++t)
      o[t] = wmma_bf16(ap, load_b_frag(Vb + (t * 16) * 32, 32, lane), o[t]);
  };

  for (int j0 = 0; j0 < NN - 64; j0 += 32) {
    int buf = (j0 >> 5) & 1;
    WAIT_ASYNC_2();     // this tile's 2 DMAs done; only next tile's outstanding
    __syncthreads();
    step(j0, buf);
    __syncthreads();    // all waves done reading tile[buf]
    async_tile16(kg + (size_t)(j0 + 64) * HD, buf ? kl1 : kl0);
    async_tile16(vg + j0 + 64,                buf ? vl1 : vl0);
  }
  WAIT_ASYNC_2();
  __syncthreads();
  step(NN - 64, ((NN - 64) >> 5) & 1);
  WAIT_ASYNC_0();       // drain
  __syncthreads();
  step(NN - 32, ((NN - 32) >> 5) & 1);

  float ms = msum[b];
#pragma unroll
  for (int g = 0; g < 8; ++g) {
    int tok = q0 + g + hi8;
    int R = b * NN + tok;
    float fac = 1.0f / (lrun[g] * (mWin[R] * ms + 1e-6f));
#pragma unroll
    for (int t = 0; t < 4; ++t)
      ao[(size_t)R * CC + h * HD + t * 16 + ncol] = f2bf(o[t][g] * fac);
  }
}

// ---------------- projection GEMM: out = ao @ proj_w^T + proj_b (f32 out) ----------------

__global__ __launch_bounds__(256) void k_proj(const bf16_t* __restrict__ ao,
                                              const bf16_t* __restrict__ wp,
                                              const float* __restrict__ pb,
                                              float* __restrict__ out) {
  __shared__ bf16_t Bs[2][64 * 32];
  int colBase = blockIdx.x * 64;   // over 1024
  int rowBase = blockIdx.y * 256;  // over 8192
  int wave = threadIdx.x >> 5, lane = threadIdx.x & 31;
  int waveRow = rowBase + wave * 32;
  int hi8 = (lane >> 4) << 3;
  int ncol = lane & 15;

  v8f acc[2][4] = {{{}, {}, {}, {}}, {{}, {}, {}, {}}};

  int srow = threadIdx.x >> 2;
  int sch  = (threadIdx.x & 3) * 8;
  const bf16_t* wsrc = wp + (size_t)(colBase + srow) * CC + sch;
  bf16_t* ldst0 = &Bs[0][srow * 32 + sch];
  bf16_t* ldst1 = &Bs[1][srow * 32 + sch];

  async_tile16(wsrc + 0,  ldst0);
  async_tile16(wsrc + 32, ldst1);

  auto step = [&](int k0, int buf) {
    v16bf a0 = load_a_frag(ao + (size_t)waveRow * CC + k0, CC, lane);
    v16bf a1 = load_a_frag(ao + (size_t)(waveRow + 16) * CC + k0, CC, lane);
#pragma unroll
    for (int t = 0; t < 4; ++t) {
      v16bf bfrag = load_b_frag(&Bs[buf][(t * 16) * 32], 32, lane);
      acc[0][t] = wmma_bf16(a0, bfrag, acc[0][t]);
      acc[1][t] = wmma_bf16(a1, bfrag, acc[1][t]);
    }
  };

  for (int k0 = 0; k0 < CC - 64; k0 += 32) {
    int buf = (k0 >> 5) & 1;
    WAIT_ASYNC_1();
    __syncthreads();
    step(k0, buf);
    __syncthreads();
    async_tile16(wsrc + k0 + 64, buf ? ldst1 : ldst0);
  }
  WAIT_ASYNC_1();
  __syncthreads();
  step(CC - 64, ((CC - 64) >> 5) & 1);
  WAIT_ASYNC_0();
  __syncthreads();
  step(CC - 32, ((CC - 32) >> 5) & 1);

#pragma unroll
  for (int mi = 0; mi < 2; ++mi) {
#pragma unroll
    for (int g = 0; g < 8; ++g) {
      int R = waveRow + mi * 16 + g + hi8;
#pragma unroll
      for (int t = 0; t < 4; ++t) {
        int Cc = colBase + t * 16 + ncol;
        out[(size_t)R * CC + Cc] = acc[mi][t][g] + pb[Cc];
      }
    }
  }
}

// ---------------- launch ----------------

extern "C" void kernel_launch(void* const* d_in, const int* in_sizes, int n_in,
                              void* d_out, int out_size, void* d_ws, size_t ws_size,
                              hipStream_t stream) {
  (void)in_sizes; (void)n_in; (void)out_size; (void)ws_size;
  const float* x      = (const float*)d_in[0];
  const float* mWin   = (const float*)d_in[1];
  const float* qkv_w  = (const float*)d_in[2];
  const float* proj_w = (const float*)d_in[3];
  const float* proj_b = (const float*)d_in[4];
  float* out = (float*)d_out;

  char* ws = (char*)d_ws;
  size_t off = 0;
  auto alloc = [&](size_t bytes) -> void* {
    void* p = ws + off;
    off = (off + bytes + 255) & ~(size_t)255;
    return p;
  };
  float*  msum   = (float*) alloc((size_t)BB * sizeof(float));
  bf16_t* xb     = (bf16_t*)alloc((size_t)BB * NN * CC * 2);
  bf16_t* wqkvb  = (bf16_t*)alloc((size_t)3 * CC * CC * 2);
  bf16_t* wprojb = (bf16_t*)alloc((size_t)CC * CC * 2);
  bf16_t* qb     = (bf16_t*)alloc((size_t)BB * HH * NN * HD * 2);
  bf16_t* kbuf   = (bf16_t*)alloc((size_t)BB * HH * NN * HD * 2);
  bf16_t* vt     = (bf16_t*)alloc((size_t)BB * HH * NN * HD * 2);
  bf16_t* ao     = (bf16_t*)alloc((size_t)BB * NN * CC * 2);

  k_f2bf<<<2048, 256, 0, stream>>>(x, xb, BB * NN * CC);
  k_f2bf<<<2048, 256, 0, stream>>>(qkv_w, wqkvb, 3 * CC * CC);
  k_f2bf<<<1024, 256, 0, stream>>>(proj_w, wprojb, CC * CC);
  k_masksum<<<BB, 256, 0, stream>>>(mWin, msum);
  k_qkv<<<dim3(48, 32), 256, 0, stream>>>(xb, wqkvb, mWin, qb, kbuf, vt);
  k_attn<<<1024, 256, 0, stream>>>(qb, kbuf, vt, mWin, msum, ao);
  k_proj<<<dim3(16, 32), 256, 0, stream>>>(ao, wprojb, proj_b, out);
}
